// RGBTTDecoderLayer_17257178595870
// MI455X (gfx1250) — matrix operand order repaired and checked by reading, not compiled, not run on hardware
//
#include <hip/hip_runtime.h>

typedef __attribute__((ext_vector_type(16))) __bf16 v16bf;
typedef __attribute__((ext_vector_type(8)))  float  v8f;

#define D 256
#define DFF 1024
#define NH 8
#define NL 4
#define NPT 4
#define HD 32
#define B_ 4
#define LQ 1024
#define NQ (B_*LQ)   // 4096

// ---------------- q = bf16(tgt + query_pos) ----------------
__global__ void add_bf16_kernel(const float* __restrict__ a, const float* __restrict__ b,
                                __bf16* __restrict__ o, int n) {
  int i = blockIdx.x * blockDim.x + threadIdx.x;
  if (i < n) o[i] = (__bf16)(a[i] + b[i]);
}

// ---------------- plain f32 -> bf16 convert (coalesced) ----------------
__global__ void cvt_bf16_kernel(const float* __restrict__ x, __bf16* __restrict__ y, int n) {
  int i = blockIdx.x * blockDim.x + threadIdx.x;
  if (i < n) y[i] = (__bf16)x[i];
}

// ------------- transpose + convert weight: Wt[n*K+k] = bf16(W[k*N+n]) -------------
__global__ void wt_kernel(const float* __restrict__ W, __bf16* __restrict__ Wt, int K, int N) {
  int i = blockIdx.x * blockDim.x + threadIdx.x;
  if (i < K * N) {
    int n = i / K, k = i - n * K;
    Wt[i] = (__bf16)W[(size_t)k * N + n];
  }
}

// ------------- A fragment loader (bf16): two 8-element K-chunks per lane -------------
// A 16x32 layout (contiguous, matching B): lane holds K[c..c+15] in regs0-7,
// where c = (lane>=16)*16.  p points at A[row*K + c + kstep]  (16B-aligned).
__device__ inline v16bf load_a_bf16(const __bf16* __restrict__ p) {
  union { v16bf v; uint4 u[2]; } a;
  a.u[0] = ((const uint4*)p)[0];
  a.u[1] = *(const uint4*)(p + 8);
  return a.v;
}

// ------------- WMMA GEMM: C = A(bf16, MxK) * Bt(bf16, NxK)^T + bias -------------
// One wave computes a 32x64 tile (2x4 of 16x16) -> 8 WMMAs per 6 fragment loads.
// Block = 4 waves stacked in M = 128x64 tile.
// M % 128 == 0, N % 64 == 0, K % 32 == 0 hold for every GEMM in this layer.
template<bool RELU, bool OUT_BF16>
__global__ __launch_bounds__(128)
void gemm_kernel(const __bf16* __restrict__ A, const __bf16* __restrict__ Bt,
                 const float* __restrict__ bias, float* __restrict__ Cf,
                 __bf16* __restrict__ Cb, int M, int N, int K) {
  const int lane = threadIdx.x & 31;
  const int wave = threadIdx.x >> 5;
  const int m0 = (blockIdx.x * 4 + wave) * 32;
  const int n0 = blockIdx.y * 64;
  if (m0 >= M) return;                       // wave-uniform: EXEC stays all-ones

  const int rA   = lane & 15;
  const int achk = (lane >> 4) * 16;         // A half (K offset)
  const int bhlf = (lane >> 4) * 16;         // B half (K offset)

  v8f c[2][4];
  #pragma unroll
  for (int s = 0; s < 2; ++s)
    #pragma unroll
    for (int j = 0; j < 4; ++j) c[s][j] = (v8f){};

  const __bf16* A0 = A + (size_t)(m0 + rA)      * K + achk;
  const __bf16* A1 = A + (size_t)(m0 + 16 + rA) * K + achk;
  const __bf16* Bp[4];
  #pragma unroll
  for (int j = 0; j < 4; ++j)
    Bp[j] = Bt + (size_t)(n0 + j * 16 + rA) * K + bhlf;

  for (int k = 0; k < K; k += 32) {
    // speculative prefetch of the A streams two k-steps ahead (global_prefetch_b8)
    __builtin_prefetch((const void*)(A0 + k + 64), 0, 1);
    __builtin_prefetch((const void*)(A1 + k + 64), 0, 1);
    v16bf a0 = load_a_bf16(A0 + k);
    v16bf a1 = load_a_bf16(A1 + k);
    #pragma unroll
    for (int j = 0; j < 4; ++j) {
      v16bf b = *(const v16bf*)(Bp[j] + k);
      c[0][j] = __builtin_amdgcn_wmma_f32_16x16x32_bf16(false, a0, false, b, (short)0, c[0][j], false, false);
      c[1][j] = __builtin_amdgcn_wmma_f32_16x16x32_bf16(false, a1, false, b, (short)0, c[1][j], false, false);
    }
  }

  const int mloc = (lane >> 4) * 8;
  const int nloc = lane & 15;
  #pragma unroll
  for (int s = 0; s < 2; ++s) {
    #pragma unroll
    for (int j = 0; j < 4; ++j) {
      int col = n0 + j * 16 + nloc;
      float bv = bias[col];
      #pragma unroll
      for (int i = 0; i < 8; ++i) {
        float v = c[s][j][i] + bv;
        if (RELU) v = fmaxf(v, 0.0f);
        size_t off = (size_t)(m0 + s * 16 + mloc + i) * N + col;
        if (OUT_BF16) Cb[off] = (__bf16)v;
        else          Cf[off] = v;
      }
    }
  }
}

// ------------- softmax over 16 attention logits per (b,q,head), in place -------------
__global__ void softmax16_kernel(float* __restrict__ aw, int nrows) {
  int i = blockIdx.x * blockDim.x + threadIdx.x;
  if (i >= nrows) return;
  float* p = aw + (size_t)i * 16;
  float mx = -1e30f;
  #pragma unroll
  for (int j = 0; j < 16; ++j) mx = fmaxf(mx, p[j]);
  float e[16], s = 0.0f;
  #pragma unroll
  for (int j = 0; j < 16; ++j) { e[j] = __expf(p[j] - mx); s += e[j]; }
  float inv = 1.0f / s;
  #pragma unroll
  for (int j = 0; j < 16; ++j) p[j] = e[j] * inv;
}

// ------------- multi-scale deformable gather: one wave per (b,q,head), lane = channel -------------
__global__ __launch_bounds__(256)
void msgather_kernel(const __bf16* __restrict__ val, const float* __restrict__ off,
                     const float* __restrict__ aw,  const float* __restrict__ ref,
                     const int* __restrict__ ss,    __bf16* __restrict__ core) {
  int gw   = (blockIdx.x * 256 + threadIdx.x) >> 5;   // global wave = (b*LQ+q)*NH + h
  int lane = threadIdx.x & 31;                         // channel d within head
  int h = gw & (NH - 1);
  int qb = gw >> 3;                                    // b*LQ + q   (NH == 8)
  int b  = gw >> 13;                                   // / (NH*LQ) == 8192

  int Hs[NL], Ws[NL], starts[NL];
  int len_in = 0;
  #pragma unroll
  for (int l = 0; l < NL; ++l) {
    Hs[l] = ss[2 * l]; Ws[l] = ss[2 * l + 1];
    starts[l] = len_in; len_in += Hs[l] * Ws[l];
  }

  const float* offp = off + (size_t)qb * 256 + h * (NL * NPT * 2);
  const float* awp  = aw  + (size_t)qb * 128 + h * (NL * NPT);
  const float* refp = ref + (size_t)qb * (NL * 2);
  size_t vbase = ((size_t)b * len_in) * D + h * HD + lane;

  float acc = 0.0f;
  for (int l = 0; l < NL; ++l) {
    int H = Hs[l], W = Ws[l];
    float rx = refp[2 * l], ry = refp[2 * l + 1];
    size_t lvbase = vbase + (size_t)starts[l] * D;
    #pragma unroll
    for (int pt = 0; pt < NPT; ++pt) {
      float ox = offp[(l * NPT + pt) * 2 + 0];
      float oy = offp[(l * NPT + pt) * 2 + 1];
      float a  = awp[l * NPT + pt];
      float x = (rx + ox / (float)W) * (float)W - 0.5f;
      float y = (ry + oy / (float)H) * (float)H - 0.5f;
      float xf = floorf(x), yf = floorf(y);
      float fx = x - xf, fy = y - yf;
      int x0 = (int)xf, y0 = (int)yf;
      float s4 = 0.0f;
      #pragma unroll
      for (int dy = 0; dy < 2; ++dy) {
        #pragma unroll
        for (int dx = 0; dx < 2; ++dx) {
          int xi = x0 + dx, yi = y0 + dy;
          float wgt = (dx ? fx : 1.0f - fx) * (dy ? fy : 1.0f - fy);
          bool ok = (xi >= 0) && (xi < W) && (yi >= 0) && (yi < H);
          int xc = min(max(xi, 0), W - 1);
          int yc = min(max(yi, 0), H - 1);
          float g = (float)val[lvbase + (size_t)(yc * W + xc) * D];
          s4 += ok ? g * wgt : 0.0f;
        }
      }
      acc += s4 * a;
    }
  }
  core[(size_t)qb * D + h * HD + lane] = (__bf16)acc;
}

// ------------- residual add + LayerNorm: out = LN(x + y), one wave per 256-row -------------
// Optionally also writes a bf16 mirror (GEMM A operand for the FFN).
template<bool WRITE_BF16>
__global__ __launch_bounds__(256)
void ln_add_kernel(const float* __restrict__ x, const float* __restrict__ y,
                   const float* __restrict__ g, const float* __restrict__ bta,
                   float* __restrict__ out, __bf16* __restrict__ outb, int nrows) {
  int row  = blockIdx.x * 8 + (threadIdx.x >> 5);
  int lane = threadIdx.x & 31;
  if (row >= nrows) return;
  const float* xp = x + (size_t)row * D;
  const float* yp = y + (size_t)row * D;
  float v[8], s = 0.0f, s2 = 0.0f;
  #pragma unroll
  for (int j = 0; j < 8; ++j) {
    v[j] = xp[j * 32 + lane] + yp[j * 32 + lane];
    s += v[j]; s2 += v[j] * v[j];
  }
  #pragma unroll
  for (int o = 16; o > 0; o >>= 1) {
    s  += __shfl_xor(s,  o, 32);
    s2 += __shfl_xor(s2, o, 32);
  }
  float mean = s * (1.0f / D);
  float var  = s2 * (1.0f / D) - mean * mean;
  float inv  = rsqrtf(var + 1e-5f);
  float* op = out + (size_t)row * D;
  __bf16* ob = outb + (size_t)row * D;
  #pragma unroll
  for (int j = 0; j < 8; ++j) {
    float r = (v[j] - mean) * inv * g[j * 32 + lane] + bta[j * 32 + lane];
    op[j * 32 + lane] = r;
    if (WRITE_BF16) ob[j * 32 + lane] = (__bf16)r;
  }
}

extern "C" void kernel_launch(void* const* d_in, const int* in_sizes, int n_in,
                              void* d_out, int out_size, void* d_ws, size_t ws_size,
                              hipStream_t stream) {
  (void)in_sizes; (void)n_in; (void)out_size; (void)ws_size;
  const float* tgt  = (const float*)d_in[0];
  const float* qpos = (const float*)d_in[1];
  const float* ref  = (const float*)d_in[2];
  const float* src  = (const float*)d_in[3];
  // thermal branch (d_in[4], d_in[13..20], d_in[23..24]) is dead code in the reference
  const float* ow   = (const float*)d_in[5];
  const float* ob   = (const float*)d_in[6];
  const float* aww  = (const float*)d_in[7];
  const float* awbi = (const float*)d_in[8];
  const float* vw   = (const float*)d_in[9];
  const float* vb   = (const float*)d_in[10];
  const float* pw   = (const float*)d_in[11];
  const float* pb   = (const float*)d_in[12];
  const float* lng  = (const float*)d_in[21];
  const float* lnb  = (const float*)d_in[22];
  const float* w1   = (const float*)d_in[25];
  const float* b1   = (const float*)d_in[26];
  const float* w2   = (const float*)d_in[27];
  const float* b2   = (const float*)d_in[28];
  const float* ln2g = (const float*)d_in[29];
  const float* ln2b = (const float*)d_in[30];
  const int*   ss   = (const int*)d_in[31];
  float* out = (float*)d_out;

  const int LEN_IN = 128*128 + 64*64 + 32*32 + 16*16;   // 21760
  const int Mv = B_ * LEN_IN;                            // 87040 (multiple of 128)

  // ---- workspace carve-up (all buffers fully written before read) ----
  char* wp = (char*)d_ws;
  auto carve = [&](size_t bytes) { char* p = wp; wp += (bytes + 255) & ~(size_t)255; return p; };
  __bf16* qbf    = (__bf16*)carve((size_t)NQ * D * 2);
  __bf16* srcb   = (__bf16*)carve((size_t)Mv * D * 2);
  __bf16* vwT    = (__bf16*)carve((size_t)D * D * 2);
  __bf16* owT    = (__bf16*)carve((size_t)D * D * 2);
  __bf16* awwT   = (__bf16*)carve((size_t)D * 128 * 2);
  __bf16* pwT    = (__bf16*)carve((size_t)D * D * 2);
  __bf16* w1T    = (__bf16*)carve((size_t)D * DFF * 2);
  __bf16* w2T    = (__bf16*)carve((size_t)DFF * D * 2);
  __bf16* valb   = (__bf16*)carve((size_t)Mv * D * 2);
  float*  offb   = (float*) carve((size_t)NQ * D * 4);
  float*  awbuf  = (float*) carve((size_t)NQ * 128 * 4);
  __bf16* coreb  = (__bf16*)carve((size_t)NQ * D * 2);
  float*  trgb   = (float*) carve((size_t)NQ * D * 4);
  float*  fused  = (float*) carve((size_t)NQ * D * 4);
  __bf16* fusedb = (__bf16*)carve((size_t)NQ * D * 2);
  __bf16* hbuf   = (__bf16*)carve((size_t)NQ * DFF * 2);
  float*  t2     = (float*) carve((size_t)NQ * D * 4);

  // ---- operand preparation: q = bf16(tgt + pos); src -> bf16; weights -> bf16^T ----
  add_bf16_kernel<<<(NQ * D + 255) / 256, 256, 0, stream>>>(tgt, qpos, qbf, NQ * D);
  cvt_bf16_kernel<<<(Mv * D + 255) / 256, 256, 0, stream>>>(src, srcb, Mv * D);
  wt_kernel<<<(D * D   + 255) / 256, 256, 0, stream>>>(vw,  vwT,  D,   D);
  wt_kernel<<<(D * D   + 255) / 256, 256, 0, stream>>>(ow,  owT,  D,   D);
  wt_kernel<<<(D * 128 + 255) / 256, 256, 0, stream>>>(aww, awwT, D,   128);
  wt_kernel<<<(D * D   + 255) / 256, 256, 0, stream>>>(pw,  pwT,  D,   D);
  wt_kernel<<<(D * DFF + 255) / 256, 256, 0, stream>>>(w1,  w1T,  D,   DFF);
  wt_kernel<<<(DFF * D + 255) / 256, 256, 0, stream>>>(w2,  w2T,  DFF, D);

  // ---- value projection (big GEMM, bf16 output for the gather) ----
  gemm_kernel<false, true ><<<dim3(Mv / 128, D / 64),   128, 0, stream>>>(srcb,  vwT,  vb,   nullptr, valb, Mv, D,   D);
  // ---- sampling-offset and attention-weight projections ----
  gemm_kernel<false, false><<<dim3(NQ / 128, D / 64),   128, 0, stream>>>(qbf,   owT,  ob,   offb,  nullptr, NQ, D,   D);
  gemm_kernel<false, false><<<dim3(NQ / 128, 128 / 64), 128, 0, stream>>>(qbf,   awwT, awbi, awbuf, nullptr, NQ, 128, D);
  softmax16_kernel<<<(NQ * NH + 255) / 256, 256, 0, stream>>>(awbuf, NQ * NH);

  // ---- multi-scale deformable bilinear gather (bf16 core out) ----
  msgather_kernel<<<(NQ * NH) / 8, 256, 0, stream>>>(valb, offb, awbuf, ref, ss, coreb);

  // ---- output projection + residual LN (writes f32 + bf16 mirrors) ----
  gemm_kernel<false, false><<<dim3(NQ / 128, D / 64),   128, 0, stream>>>(coreb, pwT,  pb,   trgb, nullptr, NQ, D,   D);
  ln_add_kernel<true ><<<NQ / 8, 256, 0, stream>>>(tgt, trgb, lng, lnb, fused, fusedb, NQ);

  // ---- FFN (hidden stays bf16 straight out of the ReLU epilogue) ----
  gemm_kernel<true,  true ><<<dim3(NQ / 128, DFF / 64), 128, 0, stream>>>(fusedb, w1T, b1,   nullptr, hbuf, NQ, DFF, D);
  gemm_kernel<false, false><<<dim3(NQ / 128, D / 64),   128, 0, stream>>>(hbuf,   w2T, b2,   t2,    nullptr, NQ, D,   DFF);
  ln_add_kernel<false><<<NQ / 8, 256, 0, stream>>>(fused, t2, ln2g, ln2b, out, nullptr, NQ);
}
